// LlamaModel_55035710931217
// MI455X (gfx1250) — compile-verified, matrix operand order checked by reading, not run on hardware
//
#include <hip/hip_runtime.h>
#include <hip/hip_bf16.h>

// ---------------- model constants ----------------
#define D   2048
#define H   32
#define HKV 8
#define HD  64
#define II  8192
#define LL  2
#define TT  1024
#define EPSF 1e-5f

typedef __attribute__((ext_vector_type(16))) _Float16 v16h;
typedef __attribute__((ext_vector_type(8)))  _Float16 v8h;
typedef __attribute__((ext_vector_type(4)))  _Float16 v4h;
typedef __attribute__((ext_vector_type(8)))  float    v8f;
typedef __attribute__((ext_vector_type(4)))  float    vf4;

union BU { v16h v; v4h q[4]; };

// ---------------------------------------------------------------------------
// WMMA fragment addressing (CDNA5 ISA 7.12.2, 16-bit A 16x32 / B 32x16):
//   lane = r + 16*hib,  hib = (kk>>3)&1 ; element e: kk = (e&7)+(e&8?16:0)+8*hib
// Buffer layout: ((mt*(K/32)+kt)*32 + lane)*16 + e -> one 32B load per lane.
// ---------------------------------------------------------------------------
__device__ __forceinline__ int frag_off(int Kdim, int m, int k) {
  int mt = m >> 4, r = m & 15;
  int kt = k >> 5, kk = k & 31;
  int hib = (kk >> 3) & 1;
  int e = (kk & 7) + ((kk & 16) ? 8 : 0);
  int lane = r + (hib << 4);
  return ((mt * (Kdim >> 5) + kt) * 32 + lane) * 16 + e;
}

#define WMMA_F16(a, b, c) \
  __builtin_amdgcn_wmma_f32_16x16x32_f16(false, (a), false, (b), (short)0, (c), false, false)

// Split workgroup barrier (CDNA5): signal only needs DScnt drained (LDS stores
// visible to sibling waves); global prefetch clauses are issued in the
// signal->wait gap so the barrier never drains LOADcnt.
#define BAR_SIG()  asm volatile("s_wait_dscnt 0x0\n\ts_barrier_signal -1" ::: "memory")
#define BAR_WAIT() asm volatile("s_barrier_wait -1" ::: "memory")

// ---------------- embedding gather ----------------
__global__ void embed_kernel(const int* __restrict__ ids,
                             const float* __restrict__ emb,
                             float* __restrict__ h) {
  int t = blockIdx.x;
  long long row = ids[t];
  for (int d = threadIdx.x; d < D; d += blockDim.x)
    h[(size_t)t * D + d] = emb[(size_t)row * D + d];
}

// ---------------- fused residual-add + RMSNorm ----------------
// mode 0: residual = h;            x = rms(h)*w        -> f16 A-fragments
// mode 1: residual = h + residual; x = rms(residual)*w -> f16 A-fragments
// mode 2: y = rms(h + residual)*w -> fp32 output (final norm)
__global__ void rmsnorm_kernel(const float* __restrict__ hin,
                               float* __restrict__ residual,
                               const float* __restrict__ w,
                               _Float16* __restrict__ xfrag,
                               float* __restrict__ out32, int mode) {
  __shared__ float red[8];
  int t = blockIdx.x, tid = threadIdx.x;
  float vals[8];
  float ss = 0.f;
#pragma unroll
  for (int j = 0; j < 8; ++j) {
    int d = tid + j * 256;
    float r = hin[(size_t)t * D + d];
    if (mode != 0) r += residual[(size_t)t * D + d];
    if (mode != 2) residual[(size_t)t * D + d] = r;
    vals[j] = r;
    ss += r * r;
  }
  ss += __shfl_xor(ss, 1, 32);  ss += __shfl_xor(ss, 2, 32);
  ss += __shfl_xor(ss, 4, 32);  ss += __shfl_xor(ss, 8, 32);
  ss += __shfl_xor(ss, 16, 32);
  if ((tid & 31) == 0) red[tid >> 5] = ss;
  __syncthreads();
  float tot = 0.f;
#pragma unroll
  for (int i = 0; i < 8; ++i) tot += red[i];
  float inv = rsqrtf(tot * (1.0f / (float)D) + EPSF);
#pragma unroll
  for (int j = 0; j < 8; ++j) {
    int d = tid + j * 256;
    float o = vals[j] * inv * w[d];
    if (mode == 2) out32[(size_t)t * D + d] = o;
    else           xfrag[frag_off(D, t, d)] = (_Float16)o;
  }
}

// ---------------------------------------------------------------------------
// GEMM: C[M,N] = A(f16 frags)[M,K] x B(f32->f16)[K,N]
// Block = 8 waves, tile 16(M) x 512(N), K-chunk 32, prefetch distance 2.
// LDS half-address for chunk element (kk, n):
//   kq = kk>>2, kk2 = kk&3, p = n>>1
//   slot = p ^ ((p>>4)&15) ^ ((kq&2)<<2)
//   addr = kq*2048 + slot*8 + (n&1)*4 + kk2        (32x512 f16 = 32KB/buffer)
// ---------------------------------------------------------------------------

#define LOADA(AR, kt) do { if ((kt) < ksteps) \
    (AR) = *(const v16h*)(Abase + (size_t)(kt) * 512); } while (0)

#define LOADG(S, kt) do { if ((kt) < ksteps) { \
    _Pragma("unroll") for (int it = 0; it < 4; ++it) { \
      int qrow = it * 2 + hb; \
      _Pragma("unroll") for (int r = 0; r < 4; ++r) \
        S[it][r] = *(const vf4*)(B + (size_t)((kt) * 32 + qrow * 4 + r) * N + n0 + nn); \
    } } } while (0)

#define STOREC(S, smb) do { \
    _Pragma("unroll") for (int it = 0; it < 4; ++it) { \
      int qrow = it * 2 + hb; \
      _Pragma("unroll") for (int jp = 0; jp < 2; ++jp) { \
        v8h p; \
        _Pragma("unroll") for (int r = 0; r < 4; ++r) { \
          p[r]     = (_Float16)S[it][r][jp * 2]; \
          p[4 + r] = (_Float16)S[it][r][jp * 2 + 1]; \
        } \
        int pr = (nn >> 1) + jp; \
        int slot = pr ^ ((pr >> 4) & 15) ^ ((qrow & 2) << 2); \
        *(v8h*)&(smb)[qrow * 2048 + slot * 8] = p; \
      } } } while (0)

#define COMPUTEC(AF, smb) do { \
    const _Float16* smp = (smb); \
    _Pragma("unroll") for (int t = 0; t < 4; ++t) { \
      int nc = wave * 64 + t * 16 + nl; \
      int prc = nc >> 1, nodd = (nc & 1) * 4; \
      BU bu; \
      _Pragma("unroll") for (int q = 0; q < 4; ++q) { \
        int kq = kqmap[q] + hi2; \
        int slot = prc ^ ((prc >> 4) & 15) ^ ((kq & 2) << 2); \
        bu.q[q] = *(const v4h*)&smp[kq * 2048 + slot * 8 + nodd]; \
      } \
      acc[t] = WMMA_F16(AF, bu.v, acc[t]); \
    } } while (0)

__global__ __launch_bounds__(256)
void gemm_f16_wmma(const _Float16* __restrict__ A,
                   const float* __restrict__ B,
                   float* __restrict__ C, int N, int Kdim) {
  __shared__ _Float16 sm[2][8 * 2048];   // 64 KB total
  constexpr int kqmap[4] = {0, 1, 4, 5};
  int tid = threadIdx.x;
  int lane = tid & 31, wave = tid >> 5;
  int nl = lane & 15;
  int hi2 = (lane >> 4) << 1;
  int hb = tid >> 7;            // producer row-half
  int nn = (tid & 127) * 4;     // producer column base
  int n0 = blockIdx.x * 512;
  int mt = blockIdx.y;
  int ksteps = Kdim >> 5;
  const _Float16* Abase = A + ((size_t)mt * (Kdim >> 5)) * 512 + lane * 16;

  v8f acc[4] = {};
  vf4 s0[4][4], s1[4][4];
  v16h a0 = {}, a1 = {};

  // prologue: chunks 0 and 1 in flight
  LOADA(a0, 0);
  LOADG(s0, 0);
  LOADA(a1, 1);
  LOADG(s1, 1);
  STOREC(s0, sm[0]);
  BAR_SIG();
  BAR_WAIT();

  for (int kt = 0; kt < ksteps; kt += 2) {
    COMPUTEC(a0, sm[0]);
    STOREC(s1, sm[1]);
    BAR_SIG();
    if (kt + 2 < ksteps) { LOADA(a0, kt + 2); LOADG(s0, kt + 2); }
    BAR_WAIT();
    COMPUTEC(a1, sm[1]);
    if (kt + 2 < ksteps) { STOREC(s0, sm[0]); }
    BAR_SIG();
    if (kt + 3 < ksteps) { LOADA(a1, kt + 3); LOADG(s1, kt + 3); }
    BAR_WAIT();
  }

  int rowbase = (mt << 4) + ((lane >> 4) << 3);
#pragma unroll
  for (int t = 0; t < 4; ++t) {
#pragma unroll
    for (int v = 0; v < 8; ++v)
      C[(size_t)(rowbase + v) * N + n0 + wave * 64 + t * 16 + nl] = acc[t][v];
  }
}

// ---------------- RoPE + pack Q/K/V into WMMA fragments ----------------
__global__ void rope_pack_kernel(const float* __restrict__ qkv,
                                 _Float16* __restrict__ qf,
                                 _Float16* __restrict__ kf,
                                 _Float16* __restrict__ vf) {
  int t = blockIdx.x;
  const float* row = qkv + (size_t)t * ((H + 2 * HKV) * HD);
  const float LOG_THETA = 9.210340371976184f;  // ln(10000)
  for (int p = threadIdx.x; p < H * 32; p += blockDim.x) {
    int h = p >> 5, i = p & 31;
    float f = expf(-(float)(2 * i) * (1.0f / 64.0f) * LOG_THETA) * (float)t;
    float s = sinf(f), c = cosf(f);
    float x1 = row[h * 64 + i], x2 = row[h * 64 + i + 32];
    _Float16* base = qf + (size_t)h * (TT * 64);
    base[frag_off(64, t, i)]      = (_Float16)((x1 * c - x2 * s) * 0.125f);
    base[frag_off(64, t, i + 32)] = (_Float16)((x2 * c + x1 * s) * 0.125f);
  }
  for (int p = threadIdx.x; p < HKV * 32; p += blockDim.x) {
    int h = p >> 5, i = p & 31;
    float f = expf(-(float)(2 * i) * (1.0f / 64.0f) * LOG_THETA) * (float)t;
    float s = sinf(f), c = cosf(f);
    float x1 = row[H * HD + h * 64 + i], x2 = row[H * HD + h * 64 + i + 32];
    _Float16* base = kf + (size_t)h * (TT * 64);
    base[frag_off(64, t, i)]      = (_Float16)(x1 * c - x2 * s);
    base[frag_off(64, t, i + 32)] = (_Float16)(x2 * c + x1 * s);
  }
  for (int p = threadIdx.x; p < HKV * 64; p += blockDim.x) {
    int h = p >> 6, d = p & 63;
    float x = row[(H + HKV) * HD + h * 64 + d];
    _Float16* base = vf + (size_t)h * (TT * 64);
    base[frag_off(TT, d, t)] = (_Float16)x;
  }
}

// ---------------- flash attention, one wave per (head, 16-row tile) --------
__global__ void attn_kernel(const _Float16* __restrict__ qf,
                            const _Float16* __restrict__ kf,
                            const _Float16* __restrict__ vf,
                            _Float16* __restrict__ of) {
  __shared__ _Float16 pbuf[16 * 32];
  int lane = threadIdx.x & 31;
  int mtile = blockIdx.x;
  int h = blockIdx.y;
  int kvh = h >> 2;  // H/HKV = 4
  const _Float16* qb = qf + (size_t)h * (TT * 64) + (size_t)mtile * 1024 + lane * 16;
  v16h qa0 = *(const v16h*)qb;
  v16h qa1 = *(const v16h*)(qb + 512);
  const _Float16* kb = kf + (size_t)kvh * (TT * 64);
  const _Float16* vb = vf + (size_t)kvh * (TT * 64);
  v8f o0 = {}, o1 = {}, o2 = {}, o3 = {};
  float rmax[8], rsum[8];
#pragma unroll
  for (int v = 0; v < 8; ++v) { rmax[v] = -1e30f; rsum[v] = 0.f; }
  int rowg = (mtile << 4) + ((lane >> 4) << 3);
  int colo = lane & 15;
  for (int jj = 0; jj <= mtile; jj += 2) {
    v8f s0 = {}, s1 = {};
    {
      v16h k0 = *(const v16h*)(kb + (size_t)(jj * 2 + 0) * 512 + lane * 16);
      v16h k1 = *(const v16h*)(kb + (size_t)(jj * 2 + 1) * 512 + lane * 16);
      v16h k2 = *(const v16h*)(kb + (size_t)(jj * 2 + 2) * 512 + lane * 16);
      v16h k3 = *(const v16h*)(kb + (size_t)(jj * 2 + 3) * 512 + lane * 16);
      s0 = WMMA_F16(qa0, k0, s0);  s0 = WMMA_F16(qa1, k1, s0);
      s1 = WMMA_F16(qa0, k2, s1);  s1 = WMMA_F16(qa1, k3, s1);
    }
#pragma unroll
    for (int v = 0; v < 8; ++v) {
      int r = rowg + v;
      if (((jj) << 4) + colo > r)     s0[v] = -1e30f;
      if (((jj + 1) << 4) + colo > r) s1[v] = -1e30f;
    }
#pragma unroll
    for (int v = 0; v < 8; ++v) {
      float m = fmaxf(s0[v], s1[v]);
      m = fmaxf(m, __shfl_xor(m, 1, 32));
      m = fmaxf(m, __shfl_xor(m, 2, 32));
      m = fmaxf(m, __shfl_xor(m, 4, 32));
      m = fmaxf(m, __shfl_xor(m, 8, 32));
      float mnew = fmaxf(rmax[v], m);
      float corr = expf(rmax[v] - mnew);
      float p0 = expf(s0[v] - mnew);
      float p1 = expf(s1[v] - mnew);
      float ts = p0 + p1;
      ts += __shfl_xor(ts, 1, 32);
      ts += __shfl_xor(ts, 2, 32);
      ts += __shfl_xor(ts, 4, 32);
      ts += __shfl_xor(ts, 8, 32);
      rsum[v] = rsum[v] * corr + ts;
      rmax[v] = mnew;
      o0[v] *= corr; o1[v] *= corr; o2[v] *= corr; o3[v] *= corr;
      int rl = v + ((lane >> 4) << 3);
      pbuf[rl * 32 + colo]      = (_Float16)p0;
      pbuf[rl * 32 + colo + 16] = (_Float16)p1;
    }
    __syncthreads();
    v16h pa;
    int kread = ((lane >> 4) & 1) << 3;
#pragma unroll
    for (int e = 0; e < 16; ++e) {
      int kk = (e & 7) + ((e & 8) << 1) + kread;
      pa[e] = pbuf[colo * 32 + kk];
    }
    __syncthreads();
    int j2 = jj >> 1;
    v16h v0 = *(const v16h*)(vb + ((size_t)(0 * (TT >> 5) + j2) * 32 + lane) * 16);
    v16h v1 = *(const v16h*)(vb + ((size_t)(1 * (TT >> 5) + j2) * 32 + lane) * 16);
    v16h v2 = *(const v16h*)(vb + ((size_t)(2 * (TT >> 5) + j2) * 32 + lane) * 16);
    v16h v3 = *(const v16h*)(vb + ((size_t)(3 * (TT >> 5) + j2) * 32 + lane) * 16);
    o0 = WMMA_F16(pa, v0, o0);
    o1 = WMMA_F16(pa, v1, o1);
    o2 = WMMA_F16(pa, v2, o2);
    o3 = WMMA_F16(pa, v3, o3);
  }
#pragma unroll
  for (int v = 0; v < 8; ++v) {
    float inv = 1.0f / rsum[v];
    int m = rowg + v;
    int cb = h * 64;
    of[frag_off(D, m, cb + 0 + colo)]  = (_Float16)(o0[v] * inv);
    of[frag_off(D, m, cb + 16 + colo)] = (_Float16)(o1[v] * inv);
    of[frag_off(D, m, cb + 32 + colo)] = (_Float16)(o2[v] * inv);
    of[frag_off(D, m, cb + 48 + colo)] = (_Float16)(o3[v] * inv);
  }
}

// ---------------- SiLU(gate)*up -> f16 A-fragments ----------------
__global__ void silu_mul_kernel(const float* __restrict__ gu,
                                _Float16* __restrict__ xf) {
  int t = blockIdx.x;
  const float* row = gu + (size_t)t * (2 * II);
  for (int i = threadIdx.x; i < II; i += blockDim.x) {
    float g = row[i], u = row[i + II];
    float v = g / (1.f + expf(-g)) * u;
    xf[frag_off(II, t, i)] = (_Float16)v;
  }
}

// ---------------- orchestration ----------------
extern "C" void kernel_launch(void* const* d_in, const int* in_sizes, int n_in,
                              void* d_out, int out_size, void* d_ws, size_t ws_size,
                              hipStream_t stream) {
  (void)in_sizes; (void)n_in; (void)out_size; (void)ws_size;
  const int*   ids      = (const int*)d_in[0];
  const float* emb      = (const float*)d_in[2];
  const float* w_qkv    = (const float*)d_in[3];
  const float* w_o      = (const float*)d_in[4];
  const float* w_gu     = (const float*)d_in[5];
  const float* w_down   = (const float*)d_in[6];
  const float* ln1      = (const float*)d_in[7];
  const float* ln2      = (const float*)d_in[8];
  const float* final_ln = (const float*)d_in[9];
  float* out = (float*)d_out;

  char* ws = (char*)d_ws;
  float*    residual = (float*)(ws);                         //  8 MB
  float*    hbuf     = (float*)(ws + ((size_t)8  << 20));    //  8 MB
  _Float16* x16      = (_Float16*)(ws + ((size_t)16 << 20)); //  4 MB
  float*    qkv      = (float*)(ws + ((size_t)20 << 20));    // 12 MB
  _Float16* q16      = (_Float16*)(ws + ((size_t)32 << 20)); //  4 MB
  _Float16* k16      = (_Float16*)(ws + ((size_t)36 << 20)); //  1 MB
  _Float16* v16b     = (_Float16*)(ws + ((size_t)37 << 20)); //  1 MB
  _Float16* attn16   = (_Float16*)(ws + ((size_t)38 << 20)); //  4 MB
  float*    gateup   = (float*)(ws + ((size_t)42 << 20));    // 64 MB
  _Float16* mlp16    = (_Float16*)(ws + ((size_t)106 << 20));// 16 MB

  const int NQKV = (H + 2 * HKV) * HD;  // 3072

  embed_kernel<<<TT, 256, 0, stream>>>(ids, emb, hbuf);
  for (int l = 0; l < LL; ++l) {
    rmsnorm_kernel<<<TT, 256, 0, stream>>>(hbuf, residual, ln1 + (size_t)l * D,
                                           x16, nullptr, l == 0 ? 0 : 1);
    gemm_f16_wmma<<<dim3(NQKV / 512, TT / 16), 256, 0, stream>>>(
        x16, w_qkv + (size_t)l * D * NQKV, qkv, NQKV, D);
    rope_pack_kernel<<<TT, 256, 0, stream>>>(qkv, q16, k16, v16b);
    attn_kernel<<<dim3(TT / 16, H), 32, 0, stream>>>(q16, k16, v16b, attn16);
    gemm_f16_wmma<<<dim3(D / 512, TT / 16), 256, 0, stream>>>(
        attn16, w_o + (size_t)l * (H * HD) * D, hbuf, D, H * HD);
    rmsnorm_kernel<<<TT, 256, 0, stream>>>(hbuf, residual, ln2 + (size_t)l * D,
                                           x16, nullptr, 1);
    gemm_f16_wmma<<<dim3((2 * II) / 512, TT / 16), 256, 0, stream>>>(
        x16, w_gu + (size_t)l * D * (2 * II), gateup, 2 * II, D);
    silu_mul_kernel<<<TT, 256, 0, stream>>>(gateup, mlp16);
    gemm_f16_wmma<<<dim3(D / 512, TT / 16), 256, 0, stream>>>(
        mlp16, w_down + (size_t)l * II * D, hbuf, D, II);
  }
  rmsnorm_kernel<<<TT, 256, 0, stream>>>(hbuf, residual, final_ln, nullptr, out, 2);
}
